// RWKV6Block_2018634629636
// MI455X (gfx1250) — compile-verified, matrix operand order checked by reading, not run on hardware
//
#include <hip/hip_runtime.h>
#include <hip/hip_bf16.h>
#include <cstdint>

typedef __attribute__((ext_vector_type(16))) __bf16 v16bf;
typedef __attribute__((ext_vector_type(8)))  __bf16 v8bf;
typedef __attribute__((ext_vector_type(8)))  float  v8f;

#define DIM   1024
#define NH    16
#define HD    64
#define BATCH 4
#define TLEN  4096
#define MTOT  (BATCH * TLEN)   // 16384

// ---------------------------------------------------------------------------
// CDNA5 async global->LDS copy (ASYNCcnt-tracked) + wait
// ---------------------------------------------------------------------------
__device__ __forceinline__ void async_copy_b128(unsigned lds_off, const void* gaddr) {
    asm volatile("global_load_async_to_lds_b128 %0, %1, off"
                 :: "v"(lds_off), "v"(gaddr) : "memory");
}
__device__ __forceinline__ void wait_async0() {
    asm volatile("s_wait_asynccnt 0x0" ::: "memory");
}

// ---------------------------------------------------------------------------
// fp32 -> bf16 conversion, optional per-column scale (time_mix broadcast)
// ---------------------------------------------------------------------------
__global__ __launch_bounds__(256)
void cvt_bf16_kernel(const float* __restrict__ in,
                     const float* __restrict__ colscale,   // nullable, length DIM
                     __hip_bfloat16* __restrict__ out,
                     long long n) {
    long long i = (long long)blockIdx.x * blockDim.x + threadIdx.x;
    if (i >= n) return;
    float v = in[i];
    if (colscale) v *= colscale[i & (DIM - 1)];
    out[i] = __float2bfloat16(v);
}

// ---------------------------------------------------------------------------
// bf16 WMMA GEMM:  C[M,N] = A[M,K] * W[N,K]^T  (torch Linear convention)
// Block = 128 threads (4 waves). Block tile 64x128, wave tile 32x64 (2x4 WMMA).
// K-tiles (K=32) staged global->LDS with async copies, double-buffered.
// LDS row = 32 bf16 (64B) + 16B pad = 80B stride (16B aligned, bank-staggered).
// ---------------------------------------------------------------------------
#define BLK_M 64
#define BLK_N 128
#define LDS_STRIDE 80
#define A_TILE_BYTES (BLK_M * LDS_STRIDE)          // 5120
#define B_TILE_BYTES (BLK_N * LDS_STRIDE)          // 10240
#define BUF_BYTES    (A_TILE_BYTES + B_TILE_BYTES) // 15360
#define SMEM_BYTES   (2 * BUF_BYTES)               // 30720

__device__ __forceinline__ v16bf lds_frag(const char* smem, unsigned row_byte, unsigned chunk_byte) {
    // two 16B chunks of a row: K sub-chunk at +chunk and +chunk+32 (16 elements apart)
    v8bf lo = *reinterpret_cast<const v8bf*>(smem + row_byte + chunk_byte);
    v8bf hi = *reinterpret_cast<const v8bf*>(smem + row_byte + chunk_byte + 32);
    v16bf f;
#pragma unroll
    for (int i = 0; i < 8; ++i) { f[i] = lo[i]; f[i + 8] = hi[i]; }
    return f;
}

__global__ __launch_bounds__(128)
void gemm_bf16_wmma(const __bf16* __restrict__ A,   // [M,K] row-major
                    const __bf16* __restrict__ Bw,  // [N,K] row-major (weight)
                    float* __restrict__ C,          // [M,N]
                    int M, int N, int K, int do_silu) {
    extern __shared__ __align__(16) char smem[];
    const int tid  = threadIdx.x;
    const int lane = tid & 31;
    const int wave = tid >> 5;
    const int bm   = blockIdx.x * BLK_M;
    const int bn   = blockIdx.y * BLK_N;
    const int wm   = (wave >> 1) * 32;   // 0 / 32 within block-M
    const int wn   = (wave & 1) * 64;    // 0 / 64 within block-N

    // WMMA 16-bit fragment addressing (CDNA5 ISA 16x32 layout)
    const int frow  = lane & 15;               // row (A) / col (B) within 16-tile
    const unsigned chsel = ((lane >> 4) * 8) * 2;  // byte offset of lane's K sub-chunk

    // cooperative async stage of one K-step tile: (64+128) rows x 4 chunks = 768 b128
    auto stage = [&](int buf, int k0) {
        const unsigned base = (unsigned)buf * BUF_BYTES;
#pragma unroll
        for (int ii = 0; ii < 6; ++ii) {
            const int c   = tid + ii * 128;     // 0..767, coalesced
            const int row = c >> 2;
            const int ch  = c & 3;
            const __bf16* g;
            unsigned loff;
            if (row < BLK_M) {
                g    = A + (long long)(bm + row) * K + k0 + ch * 8;
                loff = base + (unsigned)row * LDS_STRIDE + (unsigned)ch * 16;
            } else {
                g    = Bw + (long long)(bn + row - BLK_M) * K + k0 + ch * 8;
                loff = base + A_TILE_BYTES + (unsigned)(row - BLK_M) * LDS_STRIDE + (unsigned)ch * 16;
            }
            async_copy_b128(loff, g);
        }
    };

    v8f acc[2][4];
#pragma unroll
    for (int i = 0; i < 2; ++i)
#pragma unroll
        for (int j = 0; j < 4; ++j) acc[i][j] = (v8f){0,0,0,0,0,0,0,0};

    const int nk = K / 32;
    stage(0, 0);
    for (int it = 0; it < nk; ++it) {
        const int cur = it & 1;
        wait_async0();          // my async writes for buffer `cur` complete
        __syncthreads();        // all waves' writes complete / all prior reads done
        if (it + 1 < nk) stage(cur ^ 1, (it + 1) * 32);

        const unsigned abase = (unsigned)cur * BUF_BYTES;
        const unsigned bbase = abase + A_TILE_BYTES;
        v16bf fa[2], fb[4];
#pragma unroll
        for (int i = 0; i < 2; ++i)
            fa[i] = lds_frag(smem, abase + (unsigned)(wm + 16 * i + frow) * LDS_STRIDE, chsel);
#pragma unroll
        for (int j = 0; j < 4; ++j)
            fb[j] = lds_frag(smem, bbase + (unsigned)(wn + 16 * j + frow) * LDS_STRIDE, chsel);
#pragma unroll
        for (int i = 0; i < 2; ++i)
#pragma unroll
            for (int j = 0; j < 4; ++j)
                acc[i][j] = __builtin_amdgcn_wmma_f32_16x16x32_bf16(
                    false, fa[i], false, fb[j], (short)0, acc[i][j], false, false);
    }

    // C/D layout: element r -> (m = r + 8*(lane>=16), n = lane&15)
    const int col   = lane & 15;
    const int rbase = (lane >> 4) * 8;
#pragma unroll
    for (int i = 0; i < 2; ++i)
#pragma unroll
        for (int j = 0; j < 4; ++j) {
            const int m0 = bm + wm + 16 * i + rbase;
            const int n0 = bn + wn + 16 * j + col;
#pragma unroll
            for (int rr = 0; rr < 8; ++rr) {
                float x = acc[i][j][rr];
                if (do_silu) x = x / (1.f + __expf(-x));
                C[(long long)(m0 + rr) * N + n0] = x;
            }
        }
}

// ---------------------------------------------------------------------------
// Sequential RWKV scan. Grid (H, B), 256 threads.
// thread t: v-column vv = t>>2, k-group kg = t&3 (16 k-rows in regs).
// 4 lanes sharing a v-column are adjacent -> in-wave shfl_xor reduction.
// global_prefetch 16 steps ahead keeps per-step loads in WGP$.
// ---------------------------------------------------------------------------
#define PFD 16
__global__ __launch_bounds__(256)
void rwkv_scan_kernel(const float* __restrict__ kx, const float* __restrict__ vx,
                      const float* __restrict__ rx, const float* __restrict__ gx,
                      const float* __restrict__ time_decay,
                      const float* __restrict__ time_first,
                      float* __restrict__ pre,        // [B,T,DIM]
                      float* __restrict__ state_out)  // [B,H,HD,HD]
{
    const int h  = blockIdx.x;
    const int b  = blockIdx.y;
    const int kg = threadIdx.x & 3;
    const int vv = threadIdx.x >> 2;

    float dec[16], st[16];
#pragma unroll
    for (int j = 0; j < 16; ++j) {
        dec[j] = __expf(-time_decay[h * HD + kg * 16 + j]);
        st[j]  = 0.f;
    }
    const float tf = time_first[h * HD + vv];

    long long off = (long long)b * TLEN * DIM + h * HD;
    for (int t = 0; t < TLEN; ++t, off += DIM) {
        if (t + PFD < TLEN) {
            const long long p = off + (long long)PFD * DIM;
            __builtin_prefetch(kx + p + kg * 16, 0, 0);
            __builtin_prefetch(vx + p + vv, 0, 0);
            __builtin_prefetch(rx + p + vv, 0, 0);
            __builtin_prefetch(gx + p + vv, 0, 0);
        }
        const float4 k4a = *reinterpret_cast<const float4*>(kx + off + kg * 16);
        const float4 k4b = *reinterpret_cast<const float4*>(kx + off + kg * 16 + 4);
        const float4 k4c = *reinterpret_cast<const float4*>(kx + off + kg * 16 + 8);
        const float4 k4d = *reinterpret_cast<const float4*>(kx + off + kg * 16 + 12);
        const float kt[16] = {k4a.x, k4a.y, k4a.z, k4a.w, k4b.x, k4b.y, k4b.z, k4b.w,
                              k4c.x, k4c.y, k4c.z, k4c.w, k4d.x, k4d.y, k4d.z, k4d.w};
        const float vt  = vx[off + vv];
        const float rt  = rx[off + vv];
        const float gt  = gx[off + vv];
        const float kvv = kx[off + vv];

        float w0 = 0.f, w1 = 0.f, w2 = 0.f, w3 = 0.f;
#pragma unroll
        for (int j = 0; j < 16; j += 4) {
            float s0 = st[j + 0] * dec[j + 0] + kt[j + 0] * vt; st[j + 0] = s0; w0 += s0 * kt[j + 0];
            float s1 = st[j + 1] * dec[j + 1] + kt[j + 1] * vt; st[j + 1] = s1; w1 += s1 * kt[j + 1];
            float s2 = st[j + 2] * dec[j + 2] + kt[j + 2] * vt; st[j + 2] = s2; w2 += s2 * kt[j + 2];
            float s3 = st[j + 3] * dec[j + 3] + kt[j + 3] * vt; st[j + 3] = s3; w3 += s3 * kt[j + 3];
        }
        float wkv = (w0 + w1) + (w2 + w3);
        wkv += __shfl_xor(wkv, 1, 32);
        wkv += __shfl_xor(wkv, 2, 32);
        wkv += tf * kvv * vt;
        if (kg == 0) pre[off + vv] = rt * wkv * gt;   // g already silu'd
    }

    long long so = (((long long)b * NH + h) * HD + kg * 16) * HD + vv;
#pragma unroll
    for (int j = 0; j < 16; ++j) state_out[so + (long long)j * HD] = st[j];
}

// ---------------------------------------------------------------------------
// Host-side orchestration
// ---------------------------------------------------------------------------
extern "C" void kernel_launch(void* const* d_in, const int* in_sizes, int n_in,
                              void* d_out, int out_size, void* d_ws, size_t ws_size,
                              hipStream_t stream) {
    const float* x      = (const float*)d_in[0];
    const float* tmix   = (const float*)d_in[1];
    const float* tdecay = (const float*)d_in[2];
    const float* tfirst = (const float*)d_in[3];
    const float* W[5]   = {(const float*)d_in[4], (const float*)d_in[5],
                           (const float*)d_in[6], (const float*)d_in[7],
                           (const float*)d_in[8]};   // Wk,Wv,Wr,Wg,Wo

    float* out       = (float*)d_out;                       // [B,T,DIM]
    float* state_out = out + (long long)MTOT * DIM;         // [B,H,HD,HD]

    char*  ws  = (char*)d_ws;
    size_t off = 0;
    auto carve = [&](size_t bytes) {
        void* p = ws + off;
        off += (bytes + 255) & ~(size_t)255;
        return p;
    };
    __hip_bfloat16* xb = (__hip_bfloat16*)carve((size_t)MTOT * DIM * 2); // reused for pre-bf16
    __hip_bfloat16* Wb[5];
    for (int i = 0; i < 5; ++i) Wb[i] = (__hip_bfloat16*)carve((size_t)DIM * DIM * 2);
    float* kbuf = (float*)carve((size_t)MTOT * DIM * 4);
    float* vbuf = (float*)carve((size_t)MTOT * DIM * 4);
    float* rbuf = (float*)carve((size_t)MTOT * DIM * 4);
    float* gbuf = (float*)carve((size_t)MTOT * DIM * 4);
    float* pbuf = (float*)carve((size_t)MTOT * DIM * 4);

    const long long nx = (long long)MTOT * DIM;
    const long long nw = (long long)DIM * DIM;

    // 1) convert x*time_mix and the five weights to bf16
    cvt_bf16_kernel<<<(int)((nx + 255) / 256), 256, 0, stream>>>(x, tmix, xb, nx);
    for (int i = 0; i < 5; ++i)
        cvt_bf16_kernel<<<(int)((nw + 255) / 256), 256, 0, stream>>>(W[i], nullptr, Wb[i], nw);

    // 2) four projection GEMMs (SiLU fused into g)
    dim3 gg(MTOT / BLK_M, DIM / BLK_N);
    gemm_bf16_wmma<<<gg, 128, SMEM_BYTES, stream>>>((const __bf16*)xb, (const __bf16*)Wb[0],
                                                    kbuf, MTOT, DIM, DIM, 0);
    gemm_bf16_wmma<<<gg, 128, SMEM_BYTES, stream>>>((const __bf16*)xb, (const __bf16*)Wb[1],
                                                    vbuf, MTOT, DIM, DIM, 0);
    gemm_bf16_wmma<<<gg, 128, SMEM_BYTES, stream>>>((const __bf16*)xb, (const __bf16*)Wb[2],
                                                    rbuf, MTOT, DIM, DIM, 0);
    gemm_bf16_wmma<<<gg, 128, SMEM_BYTES, stream>>>((const __bf16*)xb, (const __bf16*)Wb[3],
                                                    gbuf, MTOT, DIM, DIM, 1);

    // 3) sequential scan -> pre-activations + final state
    rwkv_scan_kernel<<<dim3(NH, BATCH), 256, 0, stream>>>(kbuf, vbuf, rbuf, gbuf,
                                                          tdecay, tfirst, pbuf, state_out);

    // 4) output projection (reuse xb as bf16 staging for pre)
    cvt_bf16_kernel<<<(int)((nx + 255) / 256), 256, 0, stream>>>(pbuf, nullptr, xb, nx);
    gemm_bf16_wmma<<<gg, 128, SMEM_BYTES, stream>>>((const __bf16*)xb, (const __bf16*)Wb[4],
                                                    out, MTOT, DIM, DIM, 0);
}
